// DeepSeekMoe_69432441307201
// MI455X (gfx1250) — matrix operand order repaired and checked by reading, not compiled, not run on hardware
//
#include <hip/hip_runtime.h>
#include <hip/hip_bf16.h>
#include <math.h>
#include <stdint.h>

#define TOK   4096   // B*T = 2*2048
#define DIM   1024
#define NR    16
#define NS    2
#define KSTEPS (DIM / 32)
#define NTILE  128           // N columns per block (staged in LDS)
#define JT     4             // WMMA N-subtiles per wave (16x64 strip)

typedef __attribute__((ext_vector_type(16))) __bf16 v16bf;
typedef __attribute__((ext_vector_type(8)))  __bf16 v8bf;
typedef __attribute__((ext_vector_type(8)))  float  v8f;

// ---- workspace layout (bytes) ----
#define OFF_UBF    0ull
#define OFF_WRBF   (OFF_UBF  + (size_t)TOK*DIM*2)          // 8 MB
#define OFF_WCBF   (OFF_WRBF + (size_t)NR*DIM*DIM*2)       // +32 MB
#define OFF_BSUM   (OFF_WCBF + (size_t)DIM*DIM*2)          // +2 MB
#define OFF_CNT    (OFF_BSUM + (size_t)DIM*4)
#define OFF_TIDX   (OFF_CNT  + 256)
#define OFF_TGATE  (OFF_TIDX + (size_t)NR*TOK*4)

// ---------------- weight conversion: Wr->bf16, Wc = bf16(Ws0+Ws1), bsum ----
__global__ void moe_convert_kernel(const float* __restrict__ Wr,
                                   const float* __restrict__ Ws,
                                   const float* __restrict__ bs,
                                   __bf16* __restrict__ wrbf,
                                   __bf16* __restrict__ wcbf,
                                   float* __restrict__ bsum) {
    size_t i = (size_t)blockIdx.x * blockDim.x + threadIdx.x;
    const size_t nwr = (size_t)NR * DIM * DIM;
    const size_t nwc = (size_t)DIM * DIM;
    if (i < nwr) { wrbf[i] = (__bf16)Wr[i]; return; }
    size_t j = i - nwr;
    if (j < nwc) { wcbf[j] = (__bf16)(Ws[j] + Ws[nwc + j]); return; }
    size_t k = j - nwc;
    if (k < DIM)  bsum[k] = bs[k] + bs[DIM + k];
}

// ---------------- router: sigmoid scores, top-2, expert lists, u->bf16 -----
__global__ void moe_router_kernel(const float* __restrict__ u,
                                  const float* __restrict__ cent,
                                  __bf16* __restrict__ ubf,
                                  int* __restrict__ counts,
                                  int* __restrict__ tidx,
                                  float* __restrict__ tgate) {
    const int wave = threadIdx.x >> 5;
    const int lane = threadIdx.x & 31;
    const int t = blockIdx.x * 8 + wave;         // one wave32 per token
    const float* urow = u + (size_t)t * DIM;

    float acc[NR];
#pragma unroll
    for (int r = 0; r < NR; ++r) acc[r] = 0.f;

    for (int d = lane; d < DIM; d += 32) {
        float v = urow[d];
        ubf[(size_t)t * DIM + d] = (__bf16)v;    // emit bf16 copy in same pass
#pragma unroll
        for (int r = 0; r < NR; ++r) acc[r] += v * cent[r * DIM + d];
    }
#pragma unroll
    for (int r = 0; r < NR; ++r) {
#pragma unroll
        for (int off = 16; off > 0; off >>= 1)
            acc[r] += __shfl_xor(acc[r], off, 32);
    }
    if (lane == 0) {
        float s[NR];
#pragma unroll
        for (int r = 0; r < NR; ++r) s[r] = 1.f / (1.f + expf(-acc[r]));
        // top-2, first-index tie-break (matches jax.lax.top_k)
        int b0 = 0; float v0 = s[0];
#pragma unroll
        for (int r = 1; r < NR; ++r) if (s[r] > v0) { v0 = s[r]; b0 = r; }
        int b1 = (b0 == 0) ? 1 : 0; float v1 = s[b1];
#pragma unroll
        for (int r = 0; r < NR; ++r)
            if (r != b0 && s[r] > v1) { v1 = s[r]; b1 = r; }
        int s0 = atomicAdd(&counts[b0], 1);
        tidx[b0 * TOK + s0] = t; tgate[b0 * TOK + s0] = v0;
        int s1 = atomicAdd(&counts[b1], 1);
        tidx[b1 * TOK + s1] = t; tgate[b1 * TOK + s1] = v1;
    }
}

// ---- CDNA5 async global->LDS copy (ASYNCcnt-tracked), 16B per op ---------
__device__ __forceinline__ void async_copy_b128(uint32_t lds_off, const __bf16* g) {
    asm volatile("global_load_async_to_lds_b128 %0, %1, off"
                 :: "v"(lds_off), "v"((uintptr_t)g) : "memory");
}

// stage one 128col x 32K bf16 B-slice (8KB) with 256 threads:
// thread t copies 32B: col = t>>1, K-half = t&1  -> 2 async b128 per thread
__device__ __forceinline__ void stage_b(const __bf16* __restrict__ bsrc,
                                        const __bf16* ldsdst, int tid) {
    const __bf16* g = bsrc + (size_t)(tid >> 1) * DIM + (size_t)(tid & 1) * 16;
    uint32_t l = (uint32_t)(uintptr_t)(ldsdst + tid * 16);
    async_copy_b128(l,      g);
    async_copy_b128(l + 16, g + 8);
}

// ---- A fragment (16x32 bf16): lane m holds K {base..base+15}, base = 16*(lane>>4) --
__device__ __forceinline__ v16bf load_a_frag(const __bf16* rowbase, int lane) {
    const int base = (lane >> 4) * 16;
    v8bf lo = *(const v8bf*)(rowbase + base);
    v8bf hi = *(const v8bf*)(rowbase + base + 8);
    v16bf a;
#pragma unroll
    for (int i = 0; i < 8; ++i) { a[i] = lo[i]; a[8 + i] = hi[i]; }
    return a;
}

// ---- B fragment from staged LDS slice: col is 0..127 within the slice ----
__device__ __forceinline__ v16bf lds_b_frag(const __bf16* buf, int col, int lane) {
    const __bf16* p = buf + (size_t)col * 32 + (size_t)(lane >> 4) * 16;
    return *(const v16bf*)p;
}

// ---------------- shared experts + residual + bias: full dense GEMM --------
// out[t, e] = u[t, e] + bsum[e] + sum_k ubf[t,k] * Wc[e,k]
// block = 256 thr (8 waves): 64(M) x 128(N); wave = 16(M) x 64(N), JT=4.
// B slice (128 cols x 32 K) double-buffered in LDS, shared by all 8 waves.
__global__ void moe_shared_gemm_kernel(const float* __restrict__ u,
                                       const __bf16* __restrict__ ubf,
                                       const __bf16* __restrict__ wc,
                                       const float* __restrict__ bsum,
                                       float* __restrict__ out) {
    __shared__ __bf16 bstage[2][NTILE * 32];
    const int tid   = threadIdx.x;
    const int lane  = tid & 31;
    const int wave  = tid >> 5;
    const int mn    = lane & 15;
    const int wavem = wave & 3;          // M subtile
    const int waven = wave >> 2;         // N half (0/1)
    const int m0    = blockIdx.x * 64 + wavem * 16;
    const int n0    = blockIdx.y * NTILE;
    const int nloc  = waven * 64;        // this wave's base col within slice

    const __bf16* arow = ubf + (size_t)(m0 + mn) * DIM;
    const __bf16* bsrc = wc  + (size_t)n0 * DIM;

    v8f zero = {};
    v8f acc[JT];
#pragma unroll
    for (int j = 0; j < JT; ++j) acc[j] = zero;

    stage_b(bsrc, &bstage[0][0], tid);
#pragma unroll 1
    for (int kk = 0; kk < KSTEPS; ++kk) {
        const int cur = kk & 1;
        if (kk + 1 < KSTEPS) {
            stage_b(bsrc + (kk + 1) * 32, &bstage[cur ^ 1][0], tid);
            asm volatile("s_wait_asynccnt 0x2" ::: "memory");
        } else {
            asm volatile("s_wait_asynccnt 0x0" ::: "memory");
        }
        __syncthreads();
        __builtin_prefetch(arow + kk * 32 + 64, 0, 1);
        v16bf a = load_a_frag(arow + kk * 32, lane);
        // issue all B-fragment LDS loads first, then run the WMMA chain
        v16bf bfr[JT];
#pragma unroll
        for (int j = 0; j < JT; ++j)
            bfr[j] = lds_b_frag(&bstage[cur][0], nloc + j * 16 + mn, lane);
#pragma unroll
        for (int j = 0; j < JT; ++j)
            acc[j] = __builtin_amdgcn_wmma_f32_16x16x32_bf16(
                         false, a, false, bfr[j], (short)0, acc[j], false, false);
        __syncthreads();
    }
#pragma unroll
    for (int j = 0; j < JT; ++j) {
        const int col = n0 + nloc + j * 16 + mn;
        const float bias = bsum[col];
#pragma unroll
        for (int i = 0; i < 8; ++i) {
            const int row = m0 + i + 8 * (lane >> 4);
            const size_t o = (size_t)row * DIM + col;
            out[o] = u[o] + bias + acc[j][i];
        }
    }
}

// ---------------- routed experts: gathered grouped GEMM + gated scatter ----
// block = 8 waves: 64 expert-list slots x 128 cols; same LDS B pipeline.
__global__ void moe_routed_gemm_kernel(const __bf16* __restrict__ ubf,
                                       const __bf16* __restrict__ wrbf,
                                       const float* __restrict__ br,
                                       const int* __restrict__ counts,
                                       const int* __restrict__ tidx,
                                       const float* __restrict__ tgate,
                                       float* __restrict__ out) {
    __shared__ __bf16 bstage[2][NTILE * 32];
    const int r   = blockIdx.z;
    const int cnt = counts[r];
    const int m0b = blockIdx.y * 64;
    if (m0b >= cnt) return;                      // block-uniform early exit

    const int tid   = threadIdx.x;
    const int lane  = tid & 31;
    const int wave  = tid >> 5;
    const int mn    = lane & 15;
    const int wavem = wave & 3;
    const int waven = wave >> 2;
    const int m0    = m0b + wavem * 16;
    const int n0    = blockIdx.x * NTILE;
    const int nloc  = waven * 64;

    const int slot = m0 + mn;
    const int tok  = (slot < cnt) ? tidx[r * TOK + slot] : 0;
    const __bf16* arow = ubf + (size_t)tok * DIM;
    const __bf16* bsrc = wrbf + (size_t)r * DIM * DIM + (size_t)n0 * DIM;

    v8f zero = {};
    v8f acc[JT];
#pragma unroll
    for (int j = 0; j < JT; ++j) acc[j] = zero;

    stage_b(bsrc, &bstage[0][0], tid);
#pragma unroll 1
    for (int kk = 0; kk < KSTEPS; ++kk) {
        const int cur = kk & 1;
        if (kk + 1 < KSTEPS) {
            stage_b(bsrc + (kk + 1) * 32, &bstage[cur ^ 1][0], tid);
            asm volatile("s_wait_asynccnt 0x2" ::: "memory");
        } else {
            asm volatile("s_wait_asynccnt 0x0" ::: "memory");
        }
        __syncthreads();
        __builtin_prefetch(arow + kk * 32 + 64, 0, 1);
        v16bf a = load_a_frag(arow + kk * 32, lane);
        v16bf bfr[JT];
#pragma unroll
        for (int j = 0; j < JT; ++j)
            bfr[j] = lds_b_frag(&bstage[cur][0], nloc + j * 16 + mn, lane);
#pragma unroll
        for (int j = 0; j < JT; ++j)
            acc[j] = __builtin_amdgcn_wmma_f32_16x16x32_bf16(
                         false, a, false, bfr[j], (short)0, acc[j], false, false);
        __syncthreads();
    }
#pragma unroll
    for (int j = 0; j < JT; ++j) {
        const int col = n0 + nloc + j * 16 + mn;
        const float bias = br[r * DIM + col];
#pragma unroll
        for (int i = 0; i < 8; ++i) {
            const int srow = m0 + i + 8 * (lane >> 4);
            if (srow < cnt) {
                const int tko = tidx[r * TOK + srow];
                const float g = tgate[r * TOK + srow];
                atomicAdd(&out[(size_t)tko * DIM + col], g * (acc[j][i] + bias));
            }
        }
    }
}

extern "C" void kernel_launch(void* const* d_in, const int* in_sizes, int n_in,
                              void* d_out, int out_size, void* d_ws, size_t ws_size,
                              hipStream_t stream) {
    (void)in_sizes; (void)n_in; (void)out_size; (void)ws_size;
    const float* u    = (const float*)d_in[0];   // [2,2048,1024]
    const float* cent = (const float*)d_in[1];   // [16,1024]
    const float* Wr   = (const float*)d_in[2];   // [16,1024,1024]
    const float* br   = (const float*)d_in[3];   // [16,1024]
    const float* Ws   = (const float*)d_in[4];   // [2,1024,1024]
    const float* bs   = (const float*)d_in[5];   // [2,1024]
    float* out = (float*)d_out;

    char* ws = (char*)d_ws;
    __bf16* ubf   = (__bf16*)(ws + OFF_UBF);
    __bf16* wrbf  = (__bf16*)(ws + OFF_WRBF);
    __bf16* wcbf  = (__bf16*)(ws + OFF_WCBF);
    float*  bsum  = (float*) (ws + OFF_BSUM);
    int*    cntp  = (int*)   (ws + OFF_CNT);
    int*    tidx  = (int*)   (ws + OFF_TIDX);
    float*  tgate = (float*) (ws + OFF_TGATE);

    hipMemsetAsync(cntp, 0, NR * sizeof(int), stream);

    const size_t nconv = (size_t)NR * DIM * DIM + (size_t)DIM * DIM + DIM;
    moe_convert_kernel<<<dim3((unsigned)((nconv + 255) / 256)), 256, 0, stream>>>(
        Wr, Ws, bs, wrbf, wcbf, bsum);

    moe_router_kernel<<<dim3(TOK / 8), 256, 0, stream>>>(
        u, cent, ubf, cntp, tidx, tgate);

    moe_shared_gemm_kernel<<<dim3(TOK / 64, DIM / NTILE), 256, 0, stream>>>(
        u, ubf, wcbf, bsum, out);

    moe_routed_gemm_kernel<<<dim3(DIM / NTILE, TOK / 64, NR), 256, 0, stream>>>(
        ubf, wrbf, br, cntp, tidx, tgate, out);
}
